// AutoregressiveDecoder_42743514530027
// MI455X (gfx1250) — matrix operand order repaired and checked by reading, not compiled, not run on hardware
//
#include <hip/hip_runtime.h>

// ---------------------------------------------------------------------------
// Autoregressive pointer-attention decode (batched), CDNA5 / gfx1250.
//
// Pipeline (all GEMMs via v_wmma_f32_16x16x32_f16, f32 accumulate):
//   1. proj_kernel   : emb @ W_node  -> Kh [B,N,D] f16, VT [B,H,DH,N] f16,
//                                       Lh [B,N,D] f16
//   2. ctx_kernel    : mean(emb) @ W_fixed -> ctx [B,D] f32
//   3. q_kernel      : emb[cur] @ W_step + ctx, *1/sqrt(DH) -> Qh [B,T,D] f16
//   4. attn_kernel   : scores = Qh.Kh (K=DH=32: one WMMA per 16x16 tile),
//                      masked softmax (scores live in wave accumulators),
//                      heads = probs @ V -> Hh [B,T,D] f16
//   5. glimpse_kernel: Hh @ W_out * 1/sqrt(D) -> Gh f16 (aliases Qh)
//   6. logits_kernel : Gh . Lh -> 32x512 logits in accumulators,
//                      10*tanh, mask, log-softmax -> out f32 [B,T,N]
// ---------------------------------------------------------------------------

typedef __attribute__((ext_vector_type(16))) _Float16 v16h;
typedef __attribute__((ext_vector_type(8)))  _Float16 v8h;
typedef __attribute__((ext_vector_type(8)))  float    v8f;

namespace {

constexpr int kD  = 256;
constexpr int kN  = 512;
constexpr int kT  = 512;
constexpr int kH  = 8;
constexpr int kDH = 32;
constexpr int kB  = 32;
constexpr float kNegInf   = -1e9f;
constexpr float kQScale   = 0.17677669529663689f;  // 1/sqrt(32)
constexpr float kLScale   = 0.0625f;               // 1/sqrt(256)
constexpr float kTanhClip = 10.0f;

__device__ __forceinline__ v8f wmma16(v16h a, v16h b, v8f c) {
  return __builtin_amdgcn_wmma_f32_16x16x32_f16(false, a, false, b,
                                                (short)0, c, false, false);
}

__device__ __forceinline__ v16h comb(v8h lo, v8h hi) {
  v16h r;
#pragma unroll
  for (int i = 0; i < 8; ++i) { r[i] = lo[i]; r[i + 8] = hi[i]; }
  return r;
}

// A-matrix fragment (16x32 f16). rowbase -> start of this lane's row K-window.
// Lane layout (ISA 7.12.2): lanes 0-15 carry K 0..7,16..23; lanes 16-31 +8.
__device__ __forceinline__ v16h a_frag(const _Float16* rowbase, int lane) {
  const int kb = (lane & 16) ? 8 : 0;
  return comb(*(const v8h*)(rowbase + kb), *(const v8h*)(rowbase + kb + 16));
}

// B-matrix fragment (32x16 f16). colbase -> start of this lane's column's
// K-window (K-contiguous storage). Lanes 0-15: K 0..7,16..23; lanes 16-31 +8.
__device__ __forceinline__ v16h b_frag(const _Float16* colbase, int lane) {
  const int kb = (lane & 16) ? 8 : 0;
  return comb(*(const v8h*)(colbase + kb), *(const v8h*)(colbase + kb + 16));
}

// A fragment converted on the fly from fp32 row-major global data.
__device__ __forceinline__ v16h a_frag_f32(const float* rowbase, int lane) {
  const int kb = (lane & 16) ? 8 : 0;
  float4 x0 = *(const float4*)(rowbase + kb);
  float4 x1 = *(const float4*)(rowbase + kb + 4);
  float4 y0 = *(const float4*)(rowbase + kb + 16);
  float4 y1 = *(const float4*)(rowbase + kb + 20);
  v16h r;
  r[0] = (_Float16)x0.x; r[1] = (_Float16)x0.y; r[2] = (_Float16)x0.z; r[3] = (_Float16)x0.w;
  r[4] = (_Float16)x1.x; r[5] = (_Float16)x1.y; r[6] = (_Float16)x1.z; r[7] = (_Float16)x1.w;
  r[8] = (_Float16)y0.x; r[9] = (_Float16)y0.y; r[10] = (_Float16)y0.z; r[11] = (_Float16)y0.w;
  r[12] = (_Float16)y1.x; r[13] = (_Float16)y1.y; r[14] = (_Float16)y1.z; r[15] = (_Float16)y1.w;
  return r;
}

// Reductions across a 16-lane half of a wave32 (xor stays inside the half).
__device__ __forceinline__ float half_max(float v) {
#pragma unroll
  for (int m = 1; m < 16; m <<= 1) v = fmaxf(v, __shfl_xor(v, m, 32));
  return v;
}
__device__ __forceinline__ float half_sum(float v) {
#pragma unroll
  for (int m = 1; m < 16; m <<= 1) v += __shfl_xor(v, m, 32);
  return v;
}

// ------------------------- 1. proj = emb @ W_node --------------------------
// Grid: (3D/128, N/128, B). Block: 256 threads = 8 waves (4 row x 2 col).
__global__ __launch_bounds__(256) void proj_kernel(
    const float* __restrict__ emb, const float* __restrict__ Wn,
    _Float16* __restrict__ Kh, _Float16* __restrict__ VTv,
    _Float16* __restrict__ Lh) {
  __shared__ _Float16 Bt[128][40];  // W tile, K-contiguous per column
  const int b = blockIdx.z;
  const int row0 = blockIdx.y * 128;
  const int col0 = blockIdx.x * 128;
  const int tid = (int)threadIdx.x;
  const int lane = tid & 31, w = tid >> 5;
  const int wm = w >> 1, wn = w & 1;
  const int ln = lane & 15;
  const int hi = (lane & 16) ? 8 : 0;
  const float* embB = emb + (size_t)b * kN * kD;

  v8f acc[2][4] = {};
  for (int k0 = 0; k0 < kD; k0 += 32) {
#pragma unroll
    for (int i = 0; i < 16; ++i) {  // stage Bt[n][k] = Wn[k0+k][col0+n]
      int idx = tid + i * 256;
      int k = idx >> 7, n = idx & 127;
      Bt[n][k] = (_Float16)Wn[(size_t)(k0 + k) * (3 * kD) + col0 + n];
    }
    __syncthreads();
    v16h af[2];
#pragma unroll
    for (int mt = 0; mt < 2; ++mt)
      af[mt] = a_frag_f32(embB + (size_t)(row0 + wm * 32 + mt * 16 + ln) * kD + k0, lane);
#pragma unroll
    for (int nt = 0; nt < 4; ++nt) {
      v16h bf = b_frag(&Bt[wn * 64 + nt * 16 + ln][0], lane);
      acc[0][nt] = wmma16(af[0], bf, acc[0][nt]);
      acc[1][nt] = wmma16(af[1], bf, acc[1][nt]);
    }
    __syncthreads();
  }
#pragma unroll
  for (int mt = 0; mt < 2; ++mt)
#pragma unroll
    for (int nt = 0; nt < 4; ++nt)
#pragma unroll
      for (int r = 0; r < 8; ++r) {
        int n = row0 + wm * 32 + mt * 16 + r + hi;
        int col = col0 + wn * 64 + nt * 16 + ln;
        _Float16 hv = (_Float16)acc[mt][nt][r];
        if (col < kD) {
          Kh[((size_t)b * kN + n) * kD + col] = hv;
        } else if (col < 2 * kD) {
          int c = col - kD, hh = c >> 5, dh = c & 31;
          VTv[(((size_t)b * kH + hh) * kDH + dh) * kN + n] = hv;  // V transposed
        } else {
          Lh[((size_t)b * kN + n) * kD + (col - 2 * kD)] = hv;
        }
      }
}

// ----------------- 2. graph context = mean(emb) @ W_fixed ------------------
__global__ __launch_bounds__(256) void ctx_kernel(
    const float* __restrict__ emb, const float* __restrict__ Wf,
    float* __restrict__ ctx) {
  __shared__ float mean[kD];
  const int b = blockIdx.x, tid = (int)threadIdx.x;
  const float* e = emb + (size_t)b * kN * kD;
  float s = 0.f;
  for (int n = 0; n < kN; ++n) s += e[(size_t)n * kD + tid];
  mean[tid] = s * (1.0f / kN);
  __syncthreads();
  float o = 0.f;
  for (int d = 0; d < kD; ++d) o += mean[d] * Wf[(size_t)d * kD + tid];
  ctx[(size_t)b * kD + tid] = o;
}

// --------- 3. q = (emb[cur] @ W_step + ctx) * 1/sqrt(DH) -> f16 ------------
__global__ __launch_bounds__(256) void q_kernel(
    const float* __restrict__ emb, const int* __restrict__ cur,
    const float* __restrict__ Ws, const float* __restrict__ ctx,
    _Float16* __restrict__ Qh) {
  __shared__ _Float16 Bt[128][40];
  const int b = blockIdx.z;
  const int row0 = blockIdx.y * 128;
  const int col0 = blockIdx.x * 128;
  const int tid = (int)threadIdx.x;
  const int lane = tid & 31, w = tid >> 5;
  const int wm = w >> 1, wn = w & 1;
  const int ln = lane & 15;
  const int hi = (lane & 16) ? 8 : 0;
  const float* embB = emb + (size_t)b * kN * kD;

  int node[2];
#pragma unroll
  for (int mt = 0; mt < 2; ++mt)
    node[mt] = cur[(size_t)b * kT + row0 + wm * 32 + mt * 16 + ln];

  v8f acc[2][4] = {};
  for (int k0 = 0; k0 < kD; k0 += 32) {
#pragma unroll
    for (int i = 0; i < 16; ++i) {
      int idx = tid + i * 256;
      int k = idx >> 7, n = idx & 127;
      Bt[n][k] = (_Float16)Ws[(size_t)(k0 + k) * kD + col0 + n];
    }
    __syncthreads();
    v16h af[2];
#pragma unroll
    for (int mt = 0; mt < 2; ++mt)
      af[mt] = a_frag_f32(embB + (size_t)node[mt] * kD + k0, lane);
#pragma unroll
    for (int nt = 0; nt < 4; ++nt) {
      v16h bf = b_frag(&Bt[wn * 64 + nt * 16 + ln][0], lane);
      acc[0][nt] = wmma16(af[0], bf, acc[0][nt]);
      acc[1][nt] = wmma16(af[1], bf, acc[1][nt]);
    }
    __syncthreads();
  }
#pragma unroll
  for (int mt = 0; mt < 2; ++mt)
#pragma unroll
    for (int nt = 0; nt < 4; ++nt)
#pragma unroll
      for (int r = 0; r < 8; ++r) {
        int t = row0 + wm * 32 + mt * 16 + r + hi;
        int col = col0 + wn * 64 + nt * 16 + ln;
        float v = (acc[mt][nt][r] + ctx[(size_t)b * kD + col]) * kQScale;
        Qh[((size_t)b * kT + t) * kD + col] = (_Float16)v;
      }
}

// -------------------- 4. masked multi-head attention -----------------------
// Grid: (T/32, H, B). One block computes a 32(t) x 512(n) score panel in the
// accumulators of 8 waves, does masked softmax, then heads = probs @ V.
__global__ __launch_bounds__(256) void attn_kernel(
    const _Float16* __restrict__ Qh, const _Float16* __restrict__ Kh,
    const _Float16* __restrict__ VTv, const unsigned char* __restrict__ mask,
    _Float16* __restrict__ Hh) {
  __shared__ _Float16 Lp[32][528];  // unnormalized probs (f16)
  __shared__ float part[32][4];
  __shared__ float rowmax[32];
  __shared__ float rowsum[32];
  const int b = blockIdx.z, h = blockIdx.y, t0 = blockIdx.x * 32;
  const int tid = (int)threadIdx.x;
  const int lane = tid & 31, w = tid >> 5;
  const int mt = w & 1;    // 16-row half of the panel
  const int cg = w >> 1;   // 128-column group
  const int ln = lane & 15;
  const int hi = (lane & 16) ? 8 : 0;

  // scores: K = DH = 32 -> exactly one WMMA per 16x16 tile
  v8f sc[8];
  {
    v16h aq = a_frag(Qh + (size_t)(b * kT + t0 + mt * 16 + ln) * kD + h * kDH, lane);
#pragma unroll
    for (int nt = 0; nt < 8; ++nt) {
      v16h bk = b_frag(Kh + (size_t)(b * kN + cg * 128 + nt * 16 + ln) * kD + h * kDH, lane);
      v8f z = {};
      sc[nt] = wmma16(aq, bk, z);
    }
  }
  // mask + per-row max
  float mx[8];
#pragma unroll
  for (int r = 0; r < 8; ++r) mx[r] = kNegInf;
#pragma unroll
  for (int nt = 0; nt < 8; ++nt) {
    int n = cg * 128 + nt * 16 + ln;
#pragma unroll
    for (int r = 0; r < 8; ++r) {
      int t = t0 + mt * 16 + r + hi;
      bool mk = mask[(size_t)(b * kT + t) * kN + n] != 0;
      float s = mk ? sc[nt][r] : kNegInf;
      sc[nt][r] = s;
      mx[r] = fmaxf(mx[r], s);
    }
  }
#pragma unroll
  for (int r = 0; r < 8; ++r) mx[r] = half_max(mx[r]);
  if (ln < 8) part[mt * 16 + hi + ln][cg] = mx[ln];
  __syncthreads();
  if (tid < 32)
    rowmax[tid] = fmaxf(fmaxf(part[tid][0], part[tid][1]),
                        fmaxf(part[tid][2], part[tid][3]));
  __syncthreads();
  // exp + row sum, probs (unnormalized) to LDS
  float sm[8];
#pragma unroll
  for (int r = 0; r < 8; ++r) sm[r] = 0.f;
#pragma unroll
  for (int nt = 0; nt < 8; ++nt) {
    int n = cg * 128 + nt * 16 + ln;
#pragma unroll
    for (int r = 0; r < 8; ++r) {
      int row = mt * 16 + r + hi;
      float p = __expf(sc[nt][r] - rowmax[row]);
      sm[r] += p;
      Lp[row][n] = (_Float16)p;
    }
  }
#pragma unroll
  for (int r = 0; r < 8; ++r) sm[r] = half_sum(sm[r]);
  if (ln < 8) part[mt * 16 + hi + ln][cg] = sm[ln];
  __syncthreads();
  if (tid < 32)
    rowsum[tid] = part[tid][0] + part[tid][1] + part[tid][2] + part[tid][3];
  __syncthreads();
  // heads = probs @ V : 32x32 output, 4 waves x 16 WMMA over K=N=512
  if (w < 4) {
    const int mo = w >> 1;  // t tile
    const int dt = w & 1;   // dh tile
    v8f hacc = {};
    const _Float16* vbase = VTv + ((size_t)(b * kH + h) * kDH + dt * 16 + ln) * kN;
#pragma unroll
    for (int ks = 0; ks < 16; ++ks) {
      v16h ap = a_frag(&Lp[mo * 16 + ln][ks * 32], lane);
      v16h bv = b_frag(vbase + ks * 32, lane);
      hacc = wmma16(ap, bv, hacc);
    }
#pragma unroll
    for (int r = 0; r < 8; ++r) {
      int row = mo * 16 + r + hi;
      int t = t0 + row;
      float v = hacc[r] / rowsum[row];
      Hh[(size_t)(b * kT + t) * kD + h * kDH + dt * 16 + ln] = (_Float16)v;
    }
  }
}

// ----------- 5. glimpse = heads @ W_out, scaled by 1/sqrt(D) ---------------
__global__ __launch_bounds__(256) void glimpse_kernel(
    const _Float16* __restrict__ Hh, const float* __restrict__ Wo,
    _Float16* __restrict__ Gh) {
  __shared__ _Float16 Bt[128][40];
  const int b = blockIdx.z;
  const int row0 = blockIdx.y * 128;
  const int col0 = blockIdx.x * 128;
  const int tid = (int)threadIdx.x;
  const int lane = tid & 31, w = tid >> 5;
  const int wm = w >> 1, wn = w & 1;
  const int ln = lane & 15;
  const int hi = (lane & 16) ? 8 : 0;

  v8f acc[2][4] = {};
  for (int k0 = 0; k0 < kD; k0 += 32) {
#pragma unroll
    for (int i = 0; i < 16; ++i) {
      int idx = tid + i * 256;
      int k = idx >> 7, n = idx & 127;
      Bt[n][k] = (_Float16)Wo[(size_t)(k0 + k) * kD + col0 + n];
    }
    __syncthreads();
    v16h af[2];
#pragma unroll
    for (int mt = 0; mt < 2; ++mt)
      af[mt] = a_frag(Hh + (size_t)(b * kT + row0 + wm * 32 + mt * 16 + ln) * kD + k0, lane);
#pragma unroll
    for (int nt = 0; nt < 4; ++nt) {
      v16h bf = b_frag(&Bt[wn * 64 + nt * 16 + ln][0], lane);
      acc[0][nt] = wmma16(af[0], bf, acc[0][nt]);
      acc[1][nt] = wmma16(af[1], bf, acc[1][nt]);
    }
    __syncthreads();
  }
#pragma unroll
  for (int mt = 0; mt < 2; ++mt)
#pragma unroll
    for (int nt = 0; nt < 4; ++nt)
#pragma unroll
      for (int r = 0; r < 8; ++r) {
        int t = row0 + wm * 32 + mt * 16 + r + hi;
        int col = col0 + wn * 64 + nt * 16 + ln;
        Gh[((size_t)b * kT + t) * kD + col] = (_Float16)(acc[mt][nt][r] * kLScale);
      }
}

// --- 6. logits = Gh . logit_k, tanh clip, mask, log-softmax -> out f32 -----
// Grid: (T/32, B). 32x512 logits live in the accumulators of 8 waves.
__global__ __launch_bounds__(256) void logits_kernel(
    const _Float16* __restrict__ Gh, const _Float16* __restrict__ Lh,
    const unsigned char* __restrict__ mask, float* __restrict__ out) {
  __shared__ float part[32][4];
  __shared__ float rowmax[32];
  __shared__ float rowlse[32];
  const int b = blockIdx.y, t0 = blockIdx.x * 32;
  const int tid = (int)threadIdx.x;
  const int lane = tid & 31, w = tid >> 5;
  const int mt = w & 1, cg = w >> 1;
  const int ln = lane & 15;
  const int hi = (lane & 16) ? 8 : 0;

  v8f sc[8] = {};
  for (int k0 = 0; k0 < kD; k0 += 32) {
    v16h ag = a_frag(Gh + (size_t)(b * kT + t0 + mt * 16 + ln) * kD + k0, lane);
#pragma unroll
    for (int nt = 0; nt < 8; ++nt) {
      v16h bl = b_frag(Lh + (size_t)(b * kN + cg * 128 + nt * 16 + ln) * kD + k0, lane);
      sc[nt] = wmma16(ag, bl, sc[nt]);
    }
  }
  float mx[8];
#pragma unroll
  for (int r = 0; r < 8; ++r) mx[r] = kNegInf;
#pragma unroll
  for (int nt = 0; nt < 8; ++nt) {
    int n = cg * 128 + nt * 16 + ln;
#pragma unroll
    for (int r = 0; r < 8; ++r) {
      int t = t0 + mt * 16 + r + hi;
      bool mk = mask[(size_t)(b * kT + t) * kN + n] != 0;
      float v = mk ? kTanhClip * tanhf(sc[nt][r]) : kNegInf;
      sc[nt][r] = v;
      mx[r] = fmaxf(mx[r], v);
    }
  }
#pragma unroll
  for (int r = 0; r < 8; ++r) mx[r] = half_max(mx[r]);
  if (ln < 8) part[mt * 16 + hi + ln][cg] = mx[ln];
  __syncthreads();
  if (tid < 32)
    rowmax[tid] = fmaxf(fmaxf(part[tid][0], part[tid][1]),
                        fmaxf(part[tid][2], part[tid][3]));
  __syncthreads();
  float sm[8];
#pragma unroll
  for (int r = 0; r < 8; ++r) sm[r] = 0.f;
#pragma unroll
  for (int nt = 0; nt < 8; ++nt)
#pragma unroll
    for (int r = 0; r < 8; ++r) {
      int row = mt * 16 + r + hi;
      sm[r] += __expf(sc[nt][r] - rowmax[row]);
    }
#pragma unroll
  for (int r = 0; r < 8; ++r) sm[r] = half_sum(sm[r]);
  if (ln < 8) part[mt * 16 + hi + ln][cg] = sm[ln];
  __syncthreads();
  if (tid < 32)
    rowlse[tid] = logf(part[tid][0] + part[tid][1] + part[tid][2] + part[tid][3]);
  __syncthreads();
#pragma unroll
  for (int nt = 0; nt < 8; ++nt) {
    int n = cg * 128 + nt * 16 + ln;
#pragma unroll
    for (int r = 0; r < 8; ++r) {
      int row = mt * 16 + r + hi;
      int t = t0 + row;
      out[(size_t)(b * kT + t) * kN + n] = sc[nt][r] - rowmax[row] - rowlse[row];
    }
  }
}

}  // namespace

extern "C" void kernel_launch(void* const* d_in, const int* in_sizes, int n_in,
                              void* d_out, int out_size, void* d_ws, size_t ws_size,
                              hipStream_t stream) {
  (void)in_sizes; (void)n_in; (void)out_size; (void)ws_size;
  const float* emb = (const float*)d_in[0];
  const int* cur = (const int*)d_in[1];
  const unsigned char* mask = (const unsigned char*)d_in[2];
  const float* Wn = (const float*)d_in[3];
  const float* Wf = (const float*)d_in[4];
  const float* Ws = (const float*)d_in[5];
  const float* Wo = (const float*)d_in[6];
  float* out = (float*)d_out;

  const size_t matH = (size_t)kB * kN * kD * sizeof(_Float16);  // 8 MiB
  char* ws = (char*)d_ws;
  _Float16* Kh  = (_Float16*)(ws + 0 * matH);
  _Float16* VTv = (_Float16*)(ws + 1 * matH);
  _Float16* Lh  = (_Float16*)(ws + 2 * matH);
  _Float16* Qh  = (_Float16*)(ws + 3 * matH);
  _Float16* Hh  = (_Float16*)(ws + 4 * matH);
  _Float16* Gh  = Qh;  // Qh is dead after attn_kernel
  float* ctx = (float*)(ws + 5 * matH);

  proj_kernel<<<dim3(6, 4, kB), 256, 0, stream>>>(emb, Wn, Kh, VTv, Lh);
  ctx_kernel<<<dim3(kB), 256, 0, stream>>>(emb, Wf, ctx);
  q_kernel<<<dim3(2, 4, kB), 256, 0, stream>>>(emb, cur, Ws, ctx, Qh);
  attn_kernel<<<dim3(kT / 32, kH, kB), 256, 0, stream>>>(Qh, Kh, VTv, mask, Hh);
  glimpse_kernel<<<dim3(2, 4, kB), 256, 0, stream>>>(Hh, Wo, Gh);
  logits_kernel<<<dim3(kT / 32, kB), 256, 0, stream>>>(Gh, Lh, mask, out);
}
